// NetEGNN_acce_44822278701382
// MI455X (gfx1250) — matrix operand order repaired and checked by reading, not compiled, not run on hardware
//
#include <hip/hip_runtime.h>

#define BS 4
#define NN 4000
#define KN 32
#define HID 64
#define TOTAL_NODES (BS * NN)

typedef __attribute__((ext_vector_type(16))) _Float16 v16h;
typedef __attribute__((ext_vector_type(8)))  _Float16 v8h;
typedef __attribute__((ext_vector_type(8)))  float    v8f;

static __device__ __forceinline__ float silu_f(float x) {
  // fast silu: x * rcp(1+exp(-x)) (v_rcp_f32 instead of full IEEE divide)
  return x * __builtin_amdgcn_rcpf(1.f + __expf(-x));
}
static __device__ __forceinline__ float safe_norm2(float x, float y) {
  float s = x * x + y * y;
  return s > 0.f ? sqrtf(s) : 0.f;
}

// A fragment (16-bit A 16x32 layout): lane 0-15 rows M=0..15, halves 0..7 -> K=k0..k0+7,
// halves 8..15 -> K=k0+16..k0+23 ; lanes 16-31 get the +8 K offset.
static __device__ __forceinline__ v16h load_a_frag(const _Float16* base, int lane,
                                                   int row0, int kbase, int stride) {
  int row = row0 + (lane & 15);
  int k0  = kbase + ((lane >> 4) << 3);
  const _Float16* p = base + row * stride + k0;
  union { v16h v; v8h half8[2]; } u;
  u.half8[0] = *(const v8h*)(p);
  u.half8[1] = *(const v8h*)(p + 16);
  return u.v;
}

// B fragment (16-bit B 32x16): lane holds column N = ntile*16 + lane%16;
// halves 0..7 -> K=k0..k0+7, halves 8..15 -> K=k0+16..k0+23; lanes 16-31 get the +8 K offset.
// Weights are packed n-major [64][kpad] so each 8-half K run of a column is contiguous.
static __device__ __forceinline__ v16h load_b_frag(const _Float16* base, int lane,
                                                   int ntile, int kbase, int kpad) {
  int col = ntile * 16 + (lane & 15);
  int k0  = kbase + ((lane >> 4) << 3);
  const _Float16* p = base + col * kpad + k0;
  union { v16h v; v8h half8[2]; } u;
  u.half8[0] = *(const v8h*)(p);
  u.half8[1] = *(const v8h*)(p + 16);
  return u.v;
}

static __device__ __forceinline__ v8f wmma16(v16h a, v16h b, v8f c) {
  return __builtin_amdgcn_wmma_f32_16x16x32_f16(false, a, false, b, (short)0, c,
                                                false, false);
}

// D[MTILES*16 x 64] = act(A[MTILES*16 x K] * B[K x 64] + bias).
// Bias hoisted to registers; A frags hoisted per M-tile; all B frags for a tile
// loaded before the WMMA chain so ds_load_b128s overlap the matrix pipe.
template <int ASTRIDE, int KCHUNKS, int MTILES>
static __device__ __forceinline__ void mlp_tile(
    const _Float16* __restrict__ A, const _Float16* __restrict__ B,
    const float* __restrict__ bias, bool do_silu,
    const float* __restrict__ rowscale, _Float16* __restrict__ out_h,
    float* __restrict__ gsum, _Float16* __restrict__ g16, long grow0, int lane) {
  const int BKPAD = KCHUNKS * 32;
  float bn[4];
#pragma unroll
  for (int nt = 0; nt < 4; ++nt) bn[nt] = bias[nt * 16 + (lane & 15)];
#pragma unroll
  for (int mt = 0; mt < MTILES; ++mt) {
    v16h af[KCHUNKS];
#pragma unroll
    for (int kc = 0; kc < KCHUNKS; ++kc)
      af[kc] = load_a_frag(A, lane, mt * 16, kc * 32, ASTRIDE);
#pragma unroll
    for (int nt = 0; nt < 4; ++nt) {
      v16h bf[KCHUNKS];
#pragma unroll
      for (int kc = 0; kc < KCHUNKS; ++kc)
        bf[kc] = load_b_frag(B, lane, nt, kc * 32, BKPAD);
      v8f acc = {};
#pragma unroll
      for (int kc = 0; kc < KCHUNKS; ++kc) acc = wmma16(af[kc], bf[kc], acc);
      int n = nt * 16 + (lane & 15);
#pragma unroll
      for (int r = 0; r < 8; ++r) {
        int m = mt * 16 + ((lane >> 4) << 3) + r;
        float x = acc[r] + bn[nt];
        if (do_silu) x = silu_f(x);
        if (rowscale) x *= rowscale[m];
        if (out_h) out_h[m * 64 + n] = (_Float16)x;
        if (gsum) {
          long o = (grow0 + m) * 64 + n;
          float y = gsum[o] + x;
          gsum[o] = y;
          if (g16) g16[o] = (_Float16)y;
        }
      }
    }
  }
}

// ---- pack: w[k][64] f32 -> dst[n][KPAD] f16 (zero pad K) ----
__global__ void pack_kernel(const float* __restrict__ w, _Float16* __restrict__ dst,
                            int K, int KPAD) {
  int i = blockIdx.x * 256 + threadIdx.x;
  if (i >= 64 * KPAD) return;
  int n = i / KPAD, k = i - n * KPAD;
  dst[n * KPAD + k] = (_Float16)(k < K ? w[k * 64 + n] : 0.f);
}

// ---- encode: h = emb(concat(encode_v(|v|), encode_a(|a|), k_emb)); copy ped ----
__global__ void encode_kernel(const float* __restrict__ pedf,
                              const float* __restrict__ kemb,
                              const float* __restrict__ evw, const float* __restrict__ evb,
                              const float* __restrict__ eaw, const float* __restrict__ eab,
                              const float* __restrict__ embw, const float* __restrict__ embb,
                              float* __restrict__ h, _Float16* __restrict__ h16,
                              float* __restrict__ ped_cur) {
  long tid = (long)blockIdx.x * 256 + threadIdx.x;
  if (tid >= (long)TOTAL_NODES * 64) return;
  long node = tid >> 6;
  int c = (int)(tid & 63);
  const float* p = pedf + node * 6;
  float nv = safe_norm2(p[2], p[3]);
  float na = safe_norm2(p[4], p[5]);
  float acc = embb[c];
#pragma unroll
  for (int j = 0; j < 8; ++j) acc += (nv * evw[j] + evb[j]) * embw[j * 64 + c];
#pragma unroll
  for (int j = 0; j < 8; ++j) acc += (na * eaw[j] + eab[j]) * embw[(8 + j) * 64 + c];
#pragma unroll
  for (int j = 0; j < 3; ++j) acc += kemb[node * 3 + j] * embw[(16 + j) * 64 + c];
  h[node * 64 + c] = acc;
  h16[node * 64 + c] = (_Float16)acc;
  if (c < 6) ped_cur[node * 6 + c] = p[c];
}

// ---- edge kernel: one node per wave (32 edges = 2 WMMA M-tiles) ----
__global__ __launch_bounds__(128) void edge_kernel(
    const float* __restrict__ ped, const _Float16* __restrict__ h16,
    const int* __restrict__ nmask, const int* __restrict__ nidx,
    const _Float16* __restrict__ fe1p, const float* __restrict__ fe1b,
    const _Float16* __restrict__ fe2p, const float* __restrict__ fe2b,
    const _Float16* __restrict__ fx1p, const float* __restrict__ fx1b,
    const float* __restrict__ fx2w, const float* __restrict__ fx2b,
    _Float16* __restrict__ m16, float* __restrict__ agg) {
  __shared__ __align__(16) _Float16 sFe1[64 * 160];
  __shared__ __align__(16) _Float16 sFe2[64 * 64];
  __shared__ __align__(16) _Float16 sFx1[64 * 64];
  __shared__ __align__(16) _Float16 sEin[4][32 * 160];
  __shared__ __align__(16) _Float16 sM1[4][32 * 64];
  __shared__ __align__(16) _Float16 sM2[4][32 * 64];
  __shared__ float sMask[4][32];

  int t = threadIdx.x;
  for (int i = t; i < 64 * 160; i += 128) sFe1[i] = fe1p[i];
  for (int i = t; i < 64 * 64; i += 128) sFe2[i] = fe2p[i];
  for (int i = t; i < 64 * 64; i += 128) sFx1[i] = fx1p[i];

  int wave = t >> 5, lane = t & 31;
  long node = (long)blockIdx.x * 4 + wave;
  int b = (int)(node / NN);
  int e = lane;
  long ebase = node * KN + e;
  int mk = nmask[ebase];
  int idx = nidx[ebase];
  float mf = mk ? 1.f : 0.f;
  const float* ps = ped + node * 6;
  const float* pn = ped + ((long)b * NN + idx) * 6;
  float rel[6];
#pragma unroll
  for (int j = 0; j < 6; ++j) rel[j] = mf * (pn[j] - ps[j]);
  float dists = safe_norm2(rel[0], rel[1]);
  float speed = safe_norm2(rel[2], rel[3]);

  // per-edge input row: [h_self(64) | mask*h_neigh(64) | d | s | 0-pad..160]
  // pure f16 16B copies -- no f32->f16 conversion on the gather path
  v8h* row8 = (v8h*)&sEin[wave][e * 160];
  const v8h* hs8 = (const v8h*)(h16 + node * 64);
  const v8h* hn8 = (const v8h*)(h16 + ((long)b * NN + idx) * 64);
  _Float16 mh = (_Float16)mf;
#pragma unroll
  for (int c = 0; c < 8; ++c) {
    row8[c] = hs8[c];
    row8[8 + c] = hn8[c] * mh;
  }
  v8h tail = {};
  tail[0] = (_Float16)dists;
  tail[1] = (_Float16)speed;
  row8[16] = tail;
  v8h z = {};
  row8[17] = z; row8[18] = z; row8[19] = z;
  sMask[wave][e] = mf;
  __syncthreads();

  // fe1: [32x160]x[160x64] -> silu -> sM1
  mlp_tile<160, 5, 2>(sEin[wave], sFe1, fe1b, true, nullptr, sM1[wave],
                      nullptr, nullptr, 0, lane);
  __syncthreads();
  // fe2 -> silu -> mask -> sM2 (= m_ij)
  mlp_tile<64, 2, 2>(sM1[wave], sFe2, fe2b, true, sMask[wave], sM2[wave],
                     nullptr, nullptr, 0, lane);
  __syncthreads();
  // fx1 -> silu -> sM1
  mlp_tile<64, 2, 2>(sM2[wave], sFx1, fx1b, true, nullptr, sM1[wave],
                     nullptr, nullptr, 0, lane);
  __syncthreads();

  // fx2: per-edge dot(64); masked weighted reduction of rela_xy over the wave
  float fx = fx2b[0];
#pragma unroll 8
  for (int c = 0; c < 64; ++c) fx += (float)sM1[wave][e * 64 + c] * fx2w[c];
  float cx = rel[0] * fx, cy = rel[1] * fx, cm = mf;
#pragma unroll
  for (int off = 16; off >= 1; off >>= 1) {
    cx += __shfl_xor(cx, off, 32);
    cy += __shfl_xor(cy, off, 32);
    cm += __shfl_xor(cm, off, 32);
  }
  if (lane == 0) {
    float inv = 1.f / (cm + 1e-6f);
    agg[node * 2 + 0] = cx * inv;
    agg[node * 2 + 1] = cy * inv;
  }

  // m_i = column sums of masked m_ij (f32 accumulate, stored f16 for fh1 GEMM)
  float s0 = 0.f, s1 = 0.f;
  for (int r = 0; r < 32; ++r) {
    s0 += (float)sM2[wave][r * 64 + lane];
    s1 += (float)sM2[wave][r * 64 + lane + 32];
  }
  m16[node * 64 + lane] = (_Float16)s0;
  m16[node * 64 + lane + 32] = (_Float16)s1;
}

// ---- node kernel: 16 nodes per wave ----
__global__ __launch_bounds__(128) void node_kernel(
    const float* __restrict__ ped_in, float* __restrict__ ped_out,
    float* __restrict__ h, _Float16* __restrict__ h16,
    const _Float16* __restrict__ m16, const float* __restrict__ agg,
    const _Float16* __restrict__ fa1p, const float* __restrict__ fa1b,
    const float* __restrict__ fa2w, const float* __restrict__ fa2b,
    const _Float16* __restrict__ fh1p, const float* __restrict__ fh1b,
    const _Float16* __restrict__ fh2p, const float* __restrict__ fh2b,
    float* __restrict__ out_acc) {
  __shared__ __align__(16) _Float16 sFa1[64 * 64];
  __shared__ __align__(16) _Float16 sFh1[64 * 128];
  __shared__ __align__(16) _Float16 sFh2[64 * 64];
  __shared__ __align__(16) _Float16 sX[4][16 * 128];
  __shared__ __align__(16) _Float16 sU[4][16 * 64];
  __shared__ float sFa[4][16];

  int t = threadIdx.x;
  for (int i = t; i < 64 * 64; i += 128) sFa1[i] = fa1p[i];
  for (int i = t; i < 64 * 128; i += 128) sFh1[i] = fh1p[i];
  for (int i = t; i < 64 * 64; i += 128) sFh2[i] = fh2p[i];

  int wave = t >> 5, lane = t & 31;
  long nb = (long)blockIdx.x * 64 + wave * 16;

  int rowi = lane & 15, part = lane >> 4;
  const v8h* s8 = (const v8h*)((part ? m16 : h16) + (nb + rowi) * 64);
  v8h* d8 = (v8h*)&sX[wave][rowi * 128 + part * 64];
#pragma unroll
  for (int c = 0; c < 8; ++c) d8[c] = s8[c];
  __syncthreads();

  // fa1: h part (k=0..63 of sX) -> silu -> sU
  mlp_tile<128, 2, 1>(sX[wave], sFa1, fa1b, true, nullptr, sU[wave],
                      nullptr, nullptr, 0, lane);
  __syncthreads();
  if (lane < 16) {
    float v = fa2b[0];
    for (int c = 0; c < 64; ++c) v += (float)sU[wave][lane * 64 + c] * fa2w[c];
    sFa[wave][lane] = v;
  }
  __syncthreads();
  // fh1: full [h|m_i] 128 -> silu -> sU
  mlp_tile<128, 4, 1>(sX[wave], sFh1, fh1b, true, nullptr, sU[wave],
                      nullptr, nullptr, 0, lane);
  __syncthreads();
  // fh2 -> residual add into h (and refresh f16 mirror)
  mlp_tile<64, 2, 1>(sU[wave], sFh2, fh2b, false, nullptr, nullptr,
                     h, h16, nb, lane);

  // ped integrator
  if (lane < 16) {
    long node = nb + lane;
    const float* p = ped_in + node * 6;
    float fa = sFa[wave][lane];
    float ax = fa * p[4] + agg[node * 2 + 0];
    float ay = fa * p[5] + agg[node * 2 + 1];
    float vx = p[2] + ax, vy = p[3] + ay;
    float xx = p[0] + vx, xy = p[1] + vy;
    float* q = ped_out + node * 6;
    q[0] = xx; q[1] = xy; q[2] = vx; q[3] = vy; q[4] = ax; q[5] = ay;
    if (out_acc) { out_acc[node * 2] = ax; out_acc[node * 2 + 1] = ay; }
  }
}

extern "C" void kernel_launch(void* const* d_in, const int* in_sizes, int n_in,
                              void* d_out, int out_size, void* d_ws, size_t ws_size,
                              hipStream_t stream) {
  (void)in_sizes; (void)out_size; (void)ws_size;
  if (n_in < 58) return;  // expect 4 arrays + 54 param leaves (pytree order)

  const float* pedf = (const float*)d_in[0];
  const int* nmask = (const int*)d_in[1];
  const int* nidx = (const int*)d_in[2];
  const float* kemb = (const float*)d_in[3];
  const float* emb_b = (const float*)d_in[4];
  const float* emb_w = (const float*)d_in[5];
  const float* ea_b = (const float*)d_in[6];
  const float* ea_w = (const float*)d_in[7];
  const float* ev_b = (const float*)d_in[8];
  const float* ev_w = (const float*)d_in[9];
  struct LW {
    const float *fa1b, *fa1w, *fa2b, *fa2w, *fe1b, *fe1w, *fe2b, *fe2w;
    const float *fh1b, *fh1w, *fh2b, *fh2w, *fx1b, *fx1w, *fx2b, *fx2w;
  } L[3];
  for (int l = 0; l < 3; ++l) {
    const float* const* p = (const float* const*)(d_in + 10 + l * 16);
    L[l] = {p[0], p[1], p[2],  p[3],  p[4],  p[5],  p[6],  p[7],
            p[8], p[9], p[10], p[11], p[12], p[13], p[14], p[15]};
  }

  // carve workspace (256B aligned slabs)
  char* ws = (char*)d_ws;
  auto alloc = [&](size_t bytes) {
    char* p = ws;
    ws += (bytes + 255) & ~(size_t)255;
    return p;
  };
  float* h = (float*)alloc(sizeof(float) * TOTAL_NODES * 64);
  _Float16* h16 = (_Float16*)alloc(sizeof(_Float16) * TOTAL_NODES * 64);
  float* ped_a = (float*)alloc(sizeof(float) * TOTAL_NODES * 6);
  float* ped_b = (float*)alloc(sizeof(float) * TOTAL_NODES * 6);
  _Float16* m16 = (_Float16*)alloc(sizeof(_Float16) * TOTAL_NODES * 64);
  float* agg = (float*)alloc(sizeof(float) * TOTAL_NODES * 2);
  _Float16 *fe1p[3], *fe2p[3], *fx1p[3], *fa1p[3], *fh1p[3], *fh2p[3];
  for (int l = 0; l < 3; ++l) {
    fe1p[l] = (_Float16*)alloc(sizeof(_Float16) * 64 * 160);
    fe2p[l] = (_Float16*)alloc(sizeof(_Float16) * 64 * 64);
    fx1p[l] = (_Float16*)alloc(sizeof(_Float16) * 64 * 64);
    fa1p[l] = (_Float16*)alloc(sizeof(_Float16) * 64 * 64);
    fh1p[l] = (_Float16*)alloc(sizeof(_Float16) * 64 * 128);
    fh2p[l] = (_Float16*)alloc(sizeof(_Float16) * 64 * 64);
  }

  // pack weights -> f16, n-major, K zero-padded
  for (int l = 0; l < 3; ++l) {
    struct PK { const float* w; _Float16* d; int K, KPAD; } pk[6] = {
        {L[l].fe1w, fe1p[l], 130, 160}, {L[l].fe2w, fe2p[l], 64, 64},
        {L[l].fx1w, fx1p[l], 64, 64},   {L[l].fa1w, fa1p[l], 64, 64},
        {L[l].fh1w, fh1p[l], 128, 128}, {L[l].fh2w, fh2p[l], 64, 64}};
    for (int j = 0; j < 6; ++j) {
      int n = 64 * pk[j].KPAD;
      pack_kernel<<<(n + 255) / 256, 256, 0, stream>>>(pk[j].w, pk[j].d, pk[j].K,
                                                       pk[j].KPAD);
    }
  }

  // h0 / emb + ped copy
  {
    long total = (long)TOTAL_NODES * 64;
    encode_kernel<<<(int)((total + 255) / 256), 256, 0, stream>>>(
        pedf, kemb, ev_w, ev_b, ea_w, ea_b, emb_w, emb_b, h, h16, ped_a);
  }

  float* pin = ped_a;
  float* pout = ped_b;
  for (int l = 0; l < 3; ++l) {
    edge_kernel<<<TOTAL_NODES / 4, 128, 0, stream>>>(
        pin, h16, nmask, nidx, fe1p[l], L[l].fe1b, fe2p[l], L[l].fe2b, fx1p[l],
        L[l].fx1b, L[l].fx2w, L[l].fx2b, m16, agg);
    node_kernel<<<TOTAL_NODES / 64, 128, 0, stream>>>(
        pin, pout, h, h16, m16, agg, fa1p[l], L[l].fa1b, L[l].fa2w, L[l].fa2b,
        fh1p[l], L[l].fh1b, fh2p[l], L[l].fh2b,
        (l == 2) ? (float*)d_out : nullptr);
    float* tmp = pin; pin = pout; pout = tmp;
  }
}
